// RFFCausalAttention_56367150793411
// MI455X (gfx1250) — matrix operand order repaired and checked by reading, not compiled, not run on hardware
//
#include <hip/hip_runtime.h>
#include <hip/hip_bf16.h>
#include <math.h>

// ---------------- problem constants ----------------
#define BB 2
#define TT 2048
#define CC 1024
#define HH 16
#define MM 64
#define DD 64
#define EPSV 1e-6f

typedef _Float16 h16;
typedef __attribute__((ext_vector_type(16))) _Float16 v16h;
typedef __attribute__((ext_vector_type(8)))  float    v8f;
typedef __attribute__((ext_vector_type(4)))  float    f32x4;

#define LDH   72   // padded half-stride for 64-wide K tiles (144B rows, 16B aligned)
#define LDH32 40   // padded half-stride for 32-wide K tiles (80B rows, 16B aligned)

// ---------------- gfx1250 async global->LDS copy (builtin confirmed by probe) ----
#if defined(__has_builtin)
#if __has_builtin(__builtin_amdgcn_global_load_async_to_lds_b128)
#define USE_ASYNC_CP 1
#endif
#endif
#ifndef USE_ASYNC_CP
#define USE_ASYNC_CP 0
#endif

// Exact parameter types per hipcc diagnostic:
//   param0: int __attribute__((vector_size(16))) __device__(AS1) *
//   param1: same vector type in LDS (AS3)
typedef int v4i_vs __attribute__((vector_size(16)));
typedef __attribute__((address_space(1))) v4i_vs g_v4i;
typedef __attribute__((address_space(3))) v4i_vs l_v4i;

// Copy 16 bytes (8 halves) global -> LDS. Async (ASYNCcnt) when available.
__device__ __forceinline__ void cp_b128(h16* ldst, const h16* gsrc) {
#if USE_ASYNC_CP
  __builtin_amdgcn_global_load_async_to_lds_b128((g_v4i*)gsrc, (l_v4i*)ldst, 0, 0);
#else
  *(f32x4*)ldst = *(const f32x4*)gsrc;
#endif
}
__device__ __forceinline__ void cp_wait() {
#if USE_ASYNC_CP
#if __has_builtin(__builtin_amdgcn_s_wait_asynccnt)
  __builtin_amdgcn_s_wait_asynccnt(0);
#else
  asm volatile("s_wait_asynccnt 0" ::: "memory");
#endif
#endif
}

// ---------------- WMMA fragment helpers (wave32, gfx1250 layouts) ------------
// A: 16(M) x 32(K) f16. Lane L<16: row=L, kb=0; lane L>=16: row=L-16, kb=8.
// halves 0..7 = K kb..kb+7, halves 8..15 = K kb+16..kb+23.
__device__ __forceinline__ v16h frag_a_ld(const h16* base, int lda) {
  const int lane = threadIdx.x & 31;
  const int row  = lane & 15;
  const int kb   = (lane >> 4) << 3;
  const h16* p = base + row * lda + kb;
  v16h a;
  *(f32x4*)&a         = *(const f32x4*)(p);
  *(((f32x4*)&a) + 1) = *(const f32x4*)(p + 16);
  return a;
}
// B: 32(K) x 16(N) f16, stored N-major in LDS (base + n*ldb + k).
__device__ __forceinline__ v16h frag_b_ld(const h16* base, int ldb) {
  const int lane = threadIdx.x & 31;
  const int col  = lane & 15;
  const int kb   = (lane >> 4) << 3;
  const h16* p = base + col * ldb + kb;
  v16h b;
  *(f32x4*)&b         = *(const f32x4*)(p);
  *(((f32x4*)&b) + 1) = *(const f32x4*)(p + 16);
  return b;
}
__device__ __forceinline__ v8f wmma_f16(v16h a, v16h b, v8f c) {
  return __builtin_amdgcn_wmma_f32_16x16x32_f16(false, a, false, b, (short)0, c,
                                                false, false);
}
// C/D element (lane, e): M = (lane>>4)*8 + e, N = lane & 15.

// =================================================================
// Kernel 1: QKV GEMM  x[4096,1024](f32) @ W[1024,3072](f32) + b -> q,k,v f16 [B,H,T,D]
// =================================================================
__global__ __launch_bounds__(256)
void qkv_gemm(const float* __restrict__ x, const float* __restrict__ W,
              const float* __restrict__ bias,
              h16* __restrict__ q, h16* __restrict__ k, h16* __restrict__ v) {
  __shared__ __align__(16) h16 Ah[128 * LDH32];   // [m][k]
  __shared__ __align__(16) h16 Bh[128 * LDH32];   // [n][k] (transposed)
  const int tid = threadIdx.x;
  const int wid = tid >> 5;
  const int wm = wid >> 1, wn = wid & 1;          // 4x2 waves, wave tile 32x64
  const int m0 = blockIdx.y * 128, n0 = blockIdx.x * 128;

  v8f acc[2][4];
#pragma unroll
  for (int i = 0; i < 2; ++i)
#pragma unroll
    for (int j = 0; j < 4; ++j) acc[i][j] = {};

  for (int k0 = 0; k0 < CC; k0 += 32) {
    if (k0 + 32 < CC) {   // prefetch next K tiles
      __builtin_prefetch(x + (size_t)(m0 + (tid >> 1)) * CC + k0 + 32, 0, 1);
      __builtin_prefetch(W + (size_t)(k0 + 32 + (tid >> 3)) * (3 * CC) + n0, 0, 1);
    }
#pragma unroll
    for (int i = 0; i < 4; ++i) {                 // A: 128x32 f32 -> f16
      int lin = i * 256 + tid;
      int mm = lin >> 3, kk = (lin & 7) << 2;
      f32x4 g = *(const f32x4*)(x + (size_t)(m0 + mm) * CC + k0 + kk);
      h16* d = Ah + mm * LDH32 + kk;
      d[0] = (h16)g[0]; d[1] = (h16)g[1]; d[2] = (h16)g[2]; d[3] = (h16)g[3];
    }
#pragma unroll
    for (int i = 0; i < 4; ++i) {                 // B: 32x128, store N-major
      int lin = i * 256 + tid;
      int kk = lin >> 5, nn = (lin & 31) << 2;
      f32x4 g = *(const f32x4*)(W + (size_t)(k0 + kk) * (3 * CC) + n0 + nn);
#pragma unroll
      for (int j = 0; j < 4; ++j) Bh[(nn + j) * LDH32 + kk] = (h16)g[j];
    }
    __syncthreads();

    v16h af[2], bf[4];
    af[0] = frag_a_ld(Ah + (wm * 32) * LDH32, LDH32);
    af[1] = frag_a_ld(Ah + (wm * 32 + 16) * LDH32, LDH32);
#pragma unroll
    for (int j = 0; j < 4; ++j)
      bf[j] = frag_b_ld(Bh + (wn * 64 + j * 16) * LDH32, LDH32);
#pragma unroll
    for (int i = 0; i < 2; ++i)
#pragma unroll
      for (int j = 0; j < 4; ++j) acc[i][j] = wmma_f16(af[i], bf[j], acc[i][j]);
    __syncthreads();
  }

  const int lane = tid & 31;
  const int mloc0 = (lane >> 4) << 3, nloc = lane & 15;
#pragma unroll
  for (int i = 0; i < 2; ++i)
#pragma unroll
    for (int j = 0; j < 4; ++j)
#pragma unroll
      for (int e = 0; e < 8; ++e) {
        int gm = m0 + wm * 32 + i * 16 + mloc0 + e;
        int gn = n0 + wn * 64 + j * 16 + nloc;
        float val = acc[i][j][e] + bias[gn];
        int which = gn >> 10;            // 0=q 1=k 2=v
        int c = gn & (CC - 1);
        int h = c >> 6, d = c & 63;
        int bI = gm >> 11;
        int t  = gm & (TT - 1);
        h16* dst = (which == 0) ? q : ((which == 1) ? k : v);
        dst[(((size_t)bI * HH + h) * TT + t) * DD + d] = (h16)val;
      }
}

// =================================================================
// Kernel 2: RFF  q'[bh,t,m] = sqrt(2/M) * cos( q[bh,t,:] @ omega[h] + rbias[h] )
// =================================================================
__global__ __launch_bounds__(256)
void rff_gemm(const h16* __restrict__ qin, const h16* __restrict__ kin,
              const float* __restrict__ omega, const float* __restrict__ rbias,
              h16* __restrict__ qp, h16* __restrict__ kp) {
  __shared__ __align__(16) h16 Ah[128 * LDH];   // [t][d]
  __shared__ __align__(16) h16 Bh[64 * LDH];    // [m][d] = omega^T
  const int tid = threadIdx.x;
  const int wid = tid >> 5;
  const int wm = wid >> 1, wn = wid & 1;        // wave tile 32x32
  const int bh = blockIdx.y;
  const int h  = bh & (HH - 1);
  const int t0 = blockIdx.x * 128;
  const h16* src = blockIdx.z ? kin : qin;
  h16*       dst = blockIdx.z ? kp : qp;
  const h16* xb = src + ((size_t)bh * TT + t0) * DD;

#pragma unroll
  for (int i = 0; i < 4; ++i) {                 // A: 128x64 f16, pure async copy
    int lin = i * 256 + tid;                    // 1024 x 16B chunks
    int tt = lin >> 3, dd = (lin & 7) << 3;
    cp_b128(Ah + tt * LDH + dd, xb + (size_t)tt * DD + dd);
  }
#pragma unroll
  for (int i = 0; i < 4; ++i) {                 // omega[h]: [D][M] f32 -> Bh[m][d]
    int lin = i * 256 + tid;
    int dd = lin >> 4, mm = (lin & 15) << 2;
    f32x4 g = *(const f32x4*)(omega + ((size_t)h * DD + dd) * MM + mm);
#pragma unroll
    for (int j = 0; j < 4; ++j) Bh[(mm + j) * LDH + dd] = (h16)g[j];
  }
  cp_wait();
  __syncthreads();

  v8f acc[2][2];
#pragma unroll
  for (int i = 0; i < 2; ++i)
#pragma unroll
    for (int j = 0; j < 2; ++j) acc[i][j] = {};

#pragma unroll
  for (int kk = 0; kk < DD; kk += 32) {
    v16h a0 = frag_a_ld(Ah + (wm * 32) * LDH + kk, LDH);
    v16h a1 = frag_a_ld(Ah + (wm * 32 + 16) * LDH + kk, LDH);
    v16h b0 = frag_b_ld(Bh + (wn * 32) * LDH + kk, LDH);
    v16h b1 = frag_b_ld(Bh + (wn * 32 + 16) * LDH + kk, LDH);
    acc[0][0] = wmma_f16(a0, b0, acc[0][0]);
    acc[0][1] = wmma_f16(a0, b1, acc[0][1]);
    acc[1][0] = wmma_f16(a1, b0, acc[1][0]);
    acc[1][1] = wmma_f16(a1, b1, acc[1][1]);
  }

  const float scale = 0.17677669529663687f;  // sqrt(2/64)
  const int lane = tid & 31;
  const int mloc0 = (lane >> 4) << 3, nloc = lane & 15;
#pragma unroll
  for (int i = 0; i < 2; ++i)
#pragma unroll
    for (int j = 0; j < 2; ++j)
#pragma unroll
      for (int e = 0; e < 8; ++e) {
        int t = t0 + wm * 32 + i * 16 + mloc0 + e;
        int m = wn * 32 + j * 16 + nloc;
        float val = scale * cosf(acc[i][j][e] + rbias[h * MM + m]);
        dst[((size_t)bh * TT + t) * MM + m] = (h16)val;
      }
}

// =================================================================
// Kernel 3: chunked causal linear attention, one block per (b,h).
// State S[64m x 64d] lives in persistent f32 WMMA accumulators (2 tiles/wave).
// Per 64-chunk: ctx = Q'S + tril(Q'K'^T) V ; S += K'^T V ; norm via z-dot+rowsums.
// =================================================================
__global__ __launch_bounds__(256)
void causal_linear_attn(const h16* __restrict__ qp, const h16* __restrict__ kp,
                        const h16* __restrict__ v, h16* __restrict__ attn) {
  __shared__ __align__(16) h16 Qh[64 * LDH];  // [t][m]
  __shared__ __align__(16) h16 Kh[64 * LDH];  // [t][m]  (B N-major for Q'K'^T)
  __shared__ __align__(16) h16 Kt[64 * LDH];  // [m][t]  (A for K'^T V)
  __shared__ __align__(16) h16 Vt[64 * LDH];  // [d][t]  (B N-major)
  __shared__ __align__(16) h16 Sh[64 * LDH];  // [d][m]  (B N-major, dumped state)
  __shared__ __align__(16) h16 Ph[64 * LDH];  // V-rows staging, then masked scores
  __shared__ float zbuf[64];
  __shared__ float nrm[64];

  const int tid = threadIdx.x, lane = tid & 31, wid = tid >> 5;
  const int bh = blockIdx.x;
  const int b = bh >> 4, h = bh & (HH - 1);
  const int tm  = wid & 3;            // row-tile owned by this wave
  const int td0 = (wid >> 2) * 2;     // column-tile pair base
  const int mloc0 = (lane >> 4) << 3, nloc = lane & 15;

  v8f sacc[2]; sacc[0] = {}; sacc[1] = {};
  if (tid < 64) zbuf[tid] = 0.f;

  const h16* qpb = qp + (size_t)bh * TT * MM;
  const h16* kpb = kp + (size_t)bh * TT * MM;
  const h16* vb  = v  + (size_t)bh * TT * DD;

  for (int c0 = 0; c0 < TT; c0 += 64) {
    // ---- async-stage chunk operands (f16 byte copies); V rows land in Ph
#pragma unroll
    for (int i = 0; i < 2; ++i) {
      int lin = i * 256 + tid;                  // 512 x 16B chunks per matrix
      int r = lin >> 3, c = (lin & 7) << 3;
      cp_b128(Qh + r * LDH + c, qpb + (size_t)(c0 + r) * MM + c);
      cp_b128(Kh + r * LDH + c, kpb + (size_t)(c0 + r) * MM + c);
      cp_b128(Ph + r * LDH + c, vb  + (size_t)(c0 + r) * DD + c);
    }
    // ---- dump persistent state S (f32 acc) -> Sh[d][m] f16 (register source)
#pragma unroll
    for (int j = 0; j < 2; ++j)
#pragma unroll
      for (int e = 0; e < 8; ++e) {
        int m = tm * 16 + mloc0 + e;
        int d = (td0 + j) * 16 + nloc;
        Sh[d * LDH + m] = (h16)sacc[j][e];
      }
    cp_wait();
    __syncthreads();

    // ---- LDS transposes: Kt[m][t] = Kh[t][m], Vt[d][t] = Ph[t][d]
#pragma unroll
    for (int i = 0; i < 16; ++i) {
      int lin = i * 256 + tid;
      int r = lin >> 6, c = lin & 63;
      Kt[c * LDH + r] = Kh[r * LDH + c];
      Vt[c * LDH + r] = Ph[r * LDH + c];
    }
    __syncthreads();

    // ---- GEMM1: ctx_inter = Q' @ S ; GEMM2: scores = Q' @ K'^T
    v8f cacc[2]; cacc[0] = {}; cacc[1] = {};
    v8f pacc[2]; pacc[0] = {}; pacc[1] = {};
#pragma unroll
    for (int kk = 0; kk < MM; kk += 32) {
      v16h a = frag_a_ld(Qh + (tm * 16) * LDH + kk, LDH);
      cacc[0] = wmma_f16(a, frag_b_ld(Sh + (td0 * 16) * LDH + kk, LDH), cacc[0]);
      cacc[1] = wmma_f16(a, frag_b_ld(Sh + ((td0 + 1) * 16) * LDH + kk, LDH), cacc[1]);
      pacc[0] = wmma_f16(a, frag_b_ld(Kh + (td0 * 16) * LDH + kk, LDH), pacc[0]);
      pacc[1] = wmma_f16(a, frag_b_ld(Kh + ((td0 + 1) * 16) * LDH + kk, LDH), pacc[1]);
    }
    // ---- causal mask (s <= t): overwrite Ph with masked scores
#pragma unroll
    for (int j = 0; j < 2; ++j)
#pragma unroll
      for (int e = 0; e < 8; ++e) {
        int t = tm * 16 + mloc0 + e;
        int s = (td0 + j) * 16 + nloc;
        Ph[t * LDH + s] = (h16)((s <= t) ? pacc[j][e] : 0.f);
      }
    __syncthreads();

    // ---- norm[t] = q'_t . z  +  rowsum(masked scores)
    if (tid < 64) {
      float s = 0.f;
#pragma unroll
      for (int m = 0; m < 64; ++m) s += (float)Qh[tid * LDH + m] * zbuf[m];
#pragma unroll
      for (int m = 0; m < 64; ++m) s += (float)Ph[tid * LDH + m];
      nrm[tid] = s;
    }
    __syncthreads();

    // ---- GEMM3: ctx += P @ V ; GEMM4: S += K'^T @ V (persistent accumulators)
#pragma unroll
    for (int kk = 0; kk < 64; kk += 32) {
      v16h ap = frag_a_ld(Ph + (tm * 16) * LDH + kk, LDH);
      v16h b0 = frag_b_ld(Vt + (td0 * 16) * LDH + kk, LDH);
      v16h b1 = frag_b_ld(Vt + ((td0 + 1) * 16) * LDH + kk, LDH);
      cacc[0] = wmma_f16(ap, b0, cacc[0]);
      cacc[1] = wmma_f16(ap, b1, cacc[1]);
      v16h ak = frag_a_ld(Kt + (tm * 16) * LDH + kk, LDH);
      sacc[0] = wmma_f16(ak, b0, sacc[0]);
      sacc[1] = wmma_f16(ak, b1, sacc[1]);
    }

    // ---- write normalized output chunk into [B,T,C] f16 (col = h*64 + d)
    h16* ob = attn + ((size_t)b * TT + c0) * CC + h * DD;
#pragma unroll
    for (int j = 0; j < 2; ++j)
#pragma unroll
      for (int e = 0; e < 8; ++e) {
        int t = tm * 16 + mloc0 + e;
        int d = (td0 + j) * 16 + nloc;
        ob[(size_t)t * CC + d] = (h16)(cacc[j][e] / (nrm[t] + EPSV));
      }
    __syncthreads();

    // ---- z[m] += sum_t k'_chunk[t][m]
    if (tid < 64) {
      float s = zbuf[tid];
#pragma unroll
      for (int t = 0; t < 64; ++t) s += (float)Kt[tid * LDH + t];
      zbuf[tid] = s;
    }
    __syncthreads();
  }
}

// =================================================================
// Kernel 4: out(f32) = attn[4096,1024](f16) @ W_out[1024,1024](f32) + b_out
// =================================================================
__global__ __launch_bounds__(256)
void out_gemm(const h16* __restrict__ A, const float* __restrict__ W,
              const float* __restrict__ bias, float* __restrict__ out) {
  __shared__ __align__(16) h16 Ah[128 * LDH32];
  __shared__ __align__(16) h16 Bh[128 * LDH32];
  const int tid = threadIdx.x;
  const int wid = tid >> 5;
  const int wm = wid >> 1, wn = wid & 1;
  const int m0 = blockIdx.y * 128, n0 = blockIdx.x * 128;

  v8f acc[2][4];
#pragma unroll
  for (int i = 0; i < 2; ++i)
#pragma unroll
    for (int j = 0; j < 4; ++j) acc[i][j] = {};

  for (int k0 = 0; k0 < CC; k0 += 32) {
    if (k0 + 32 < CC)
      __builtin_prefetch(W + (size_t)(k0 + 32 + (tid >> 3)) * CC + n0, 0, 1);
#pragma unroll
    for (int i = 0; i < 2; ++i) {                 // A tile: pure f16 async copy
      int lin = i * 256 + tid;                    // 512 x 16B chunks
      int mm = lin >> 2, kk = (lin & 3) << 3;
      cp_b128(Ah + mm * LDH32 + kk, A + (size_t)(m0 + mm) * CC + k0 + kk);
    }
#pragma unroll
    for (int i = 0; i < 4; ++i) {                 // B: 32x128 f32, store N-major
      int lin = i * 256 + tid;
      int kk = lin >> 5, nn = (lin & 31) << 2;
      f32x4 g = *(const f32x4*)(W + (size_t)(k0 + kk) * CC + n0 + nn);
#pragma unroll
      for (int j = 0; j < 4; ++j) Bh[(nn + j) * LDH32 + kk] = (h16)g[j];
    }
    cp_wait();
    __syncthreads();

    v16h af[2], bf[4];
    af[0] = frag_a_ld(Ah + (wm * 32) * LDH32, LDH32);
    af[1] = frag_a_ld(Ah + (wm * 32 + 16) * LDH32, LDH32);
#pragma unroll
    for (int j = 0; j < 4; ++j)
      bf[j] = frag_b_ld(Bh + (wn * 64 + j * 16) * LDH32, LDH32);
#pragma unroll
    for (int i = 0; i < 2; ++i)
#pragma unroll
      for (int j = 0; j < 4; ++j) acc[i][j] = wmma_f16(af[i], bf[j], acc[i][j]);
    __syncthreads();
  }

  const int lane = tid & 31;
  const int mloc0 = (lane >> 4) << 3, nloc = lane & 15;
#pragma unroll
  for (int i = 0; i < 2; ++i)
#pragma unroll
    for (int j = 0; j < 4; ++j)
#pragma unroll
      for (int e = 0; e < 8; ++e) {
        int gm = m0 + wm * 32 + i * 16 + mloc0 + e;
        int gn = n0 + wn * 64 + j * 16 + nloc;
        out[(size_t)gm * CC + gn] = acc[i][j][e] + bias[gn];
      }
}

// =================================================================
extern "C" void kernel_launch(void* const* d_in, const int* in_sizes, int n_in,
                              void* d_out, int out_size, void* d_ws, size_t ws_size,
                              hipStream_t stream) {
  const float* x     = (const float*)d_in[0];
  const float* W_qkv = (const float*)d_in[1];
  const float* b_qkv = (const float*)d_in[2];
  const float* W_out = (const float*)d_in[3];
  const float* b_out = (const float*)d_in[4];
  const float* omega = (const float*)d_in[5];
  const float* rbias = (const float*)d_in[6];

  const size_t NE = (size_t)BB * HH * TT * DD;   // 4,194,304 elems per buffer
  h16* q    = (h16*)d_ws;                        // all intermediates f16
  h16* k    = q + NE;
  h16* v    = k + NE;
  h16* qp   = v + NE;
  h16* kp   = qp + NE;
  h16* attn = kp + NE;

  dim3 blk(256);
  // 1) QKV projection: [4096,1024] x [1024,3072]
  qkv_gemm<<<dim3(3 * CC / 128, (BB * TT) / 128), blk, 0, stream>>>(
      x, W_qkv, b_qkv, q, k, v);
  // 2) Random-feature maps for q and k (z-dim selects q/k)
  rff_gemm<<<dim3(TT / 128, BB * HH, 2), blk, 0, stream>>>(
      q, k, omega, rbias, qp, kp);
  // 3) Chunked causal linear attention, one block per (b,h)
  causal_linear_attn<<<dim3(BB * HH), blk, 0, stream>>>(qp, kp, v, attn);
  // 4) Output projection: [4096,1024] x [1024,1024]
  out_gemm<<<dim3(CC / 128, (BB * TT) / 128), blk, 0, stream>>>(
      attn, W_out, b_out, (float*)d_out);
}
